// MSSA_80161269612898
// MI455X (gfx1250) — matrix operand order repaired and checked by reading, not compiled, not run on hardware
//
#include <hip/hip_runtime.h>
#include <hip/hip_bf16.h>

// ---------------------------------------------------------------------------
// Problem constants (from the reference)
// ---------------------------------------------------------------------------
#define BATCH   8
#define SEQ     2048
#define DIM     512          // model dim == contraction dim of attention
#define QKV_N   1536         // 3*INNER
#define ROWS    (BATCH * SEQ) // 16384

typedef __bf16 bf16_t;
typedef __attribute__((ext_vector_type(16))) __bf16 v16bf;
typedef __attribute__((ext_vector_type(8)))  __bf16 v8bf;
typedef __attribute__((ext_vector_type(8)))  float  v8f;
typedef __attribute__((ext_vector_type(4)))  int    i32x4;

// ---------------------------------------------------------------------------
// CDNA5 async global->LDS copy (ASYNCcnt-tracked), guarded by __has_builtin so
// the build is safe on toolchains without the gfx1250 async intrinsics.
// Probe feedback: param0 is AS(1) pointer-to-int4, so cast accordingly.
// ---------------------------------------------------------------------------
#if defined(__has_builtin)
#  if __has_builtin(__builtin_amdgcn_global_load_async_to_lds_b128) && \
      __has_builtin(__builtin_amdgcn_s_wait_asynccnt)
#    define USE_ASYNC_LDS 1
#  endif
#endif
#ifndef USE_ASYNC_LDS
#  define USE_ASYNC_LDS 0
#endif

#if USE_ASYNC_LDS
__device__ __forceinline__ void async_copy_b128(const void* g, void* l) {
    // generic->AS1 keeps the 64-bit value; generic LDS->AS3 truncates to the
    // 32-bit LDS offset.
    __builtin_amdgcn_global_load_async_to_lds_b128(
        (__attribute__((address_space(1))) i32x4*)(unsigned long long)g,
        (__attribute__((address_space(3))) i32x4*)(unsigned int)(unsigned long long)l,
        0, 0);
}
__device__ __forceinline__ void async_fence() {
    __builtin_amdgcn_s_wait_asynccnt(0);
}
#else
__device__ __forceinline__ void async_copy_b128(const void* g, void* l) {
    *(v8bf*)l = *(const v8bf*)g;
}
__device__ __forceinline__ void async_fence() {}
#endif

__device__ __forceinline__ v8f vzero8() {
    v8f z;
#pragma unroll
    for (int i = 0; i < 8; ++i) z[i] = 0.0f;
    return z;
}

// ---------------------------------------------------------------------------
// Core block GEMM: 128(M) x 128(N) tile, K-loop step 32, bf16 WMMA, f32 acc.
// 256 threads = 8 waves; wave (wm = wid>>1, wn = wid&1) owns 32x64 patch.
// A source is either f32 (converted to bf16 on the fly) or bf16 (async-staged).
// B source must be [n][k] row-major bf16 (contraction contiguous per row).
// ---------------------------------------------------------------------------
template<bool A_IS_F32>
__device__ __forceinline__ void gemm_block(
    const void* __restrict__ A, long lda,
    const bf16_t* __restrict__ B, long ldb,
    int K, bf16_t* As, bf16_t* Bs, v8f acc[2][4])
{
    const int tid  = threadIdx.x;
    const int ar   = tid >> 1;          // 0..127 : row of A tile / n-row of B tile
    const int kc   = (tid & 1) * 16;    // 0 or 16 : k sub-column
    const int lane = tid & 31;
    const int lr   = lane & 15;
    const int lh   = lane >> 4;
    const int wid  = tid >> 5;
    const int wm   = wid >> 1;          // 0..3
    const int wn   = wid & 1;           // 0..1

#pragma unroll
    for (int mi = 0; mi < 2; ++mi)
#pragma unroll
        for (int ni = 0; ni < 4; ++ni) acc[mi][ni] = vzero8();

    for (int kk = 0; kk < K; kk += 32) {
        // ---- stage A slab (128x32) into LDS ----
        if (A_IS_F32) {
            const float* ap = (const float*)A + (long)ar * lda + kk + kc;
#pragma unroll
            for (int i = 0; i < 4; ++i) {
                float4 f = ((const float4*)ap)[i];
                bf16_t* d = As + ar * 32 + kc + i * 4;
                d[0] = (bf16_t)f.x; d[1] = (bf16_t)f.y;
                d[2] = (bf16_t)f.z; d[3] = (bf16_t)f.w;
            }
            if (kk + 32 < K) __builtin_prefetch(ap + 32, 0, 1);
        } else {
            const bf16_t* ap = (const bf16_t*)A + (long)ar * lda + kk + kc;
            async_copy_b128(ap,     As + ar * 32 + kc);
            async_copy_b128(ap + 8, As + ar * 32 + kc + 8);
        }
        // ---- stage B slab (128 n-rows x 32 k) into LDS ----
        {
            const bf16_t* bp = B + (long)ar * ldb + kk + kc;
            async_copy_b128(bp,     Bs + ar * 32 + kc);
            async_copy_b128(bp + 8, Bs + ar * 32 + kc + 8);
        }
        async_fence();          // s_wait_asynccnt 0 : this wave's LDS writes done
        __syncthreads();        // all waves' writes visible

        // ---- B fragments: lane -> column (lr), 16 contiguous k at 16*lh ----
        v16bf bfr[4];
#pragma unroll
        for (int ni = 0; ni < 4; ++ni) {
            const bf16_t* p = Bs + (wn * 64 + ni * 16 + lr) * 32 + 16 * lh;
            bfr[ni] = *(const v16bf*)p;
        }
        // ---- A fragments: lane -> row (lr), 16 contiguous k at 16*lh ----
#pragma unroll
        for (int mi = 0; mi < 2; ++mi) {
            const bf16_t* p = As + (wm * 32 + mi * 16 + lr) * 32 + 16 * lh;
            v8bf lo = *(const v8bf*)p;
            v8bf hi = *(const v8bf*)(p + 8);
            v16bf a;
#pragma unroll
            for (int i = 0; i < 8; ++i) { a[i] = lo[i]; a[i + 8] = hi[i]; }
#pragma unroll
            for (int ni = 0; ni < 4; ++ni)
                acc[mi][ni] = __builtin_amdgcn_wmma_f32_16x16x32_bf16(
                    false, a, false, bfr[ni], (short)0, acc[mi][ni], false, false);
        }
        __syncthreads();
    }
}

// C/D fragment coords: N = lane&15 (+tile), M = j + 8*(lane>>4) (+tile)
// Calls f(r, c, val) for each of this lane's 64 result elements.
template<class F>
__device__ __forceinline__ void epilogue_for_each(const v8f acc[2][4], F f)
{
    const int lane = threadIdx.x & 31;
    const int lr = lane & 15, lh = lane >> 4;
    const int wid = threadIdx.x >> 5;
    const int wm = wid >> 1, wn = wid & 1;
#pragma unroll
    for (int mi = 0; mi < 2; ++mi)
#pragma unroll
        for (int ni = 0; ni < 4; ++ni)
#pragma unroll
            for (int j = 0; j < 8; ++j) {
                const long r = (long)wm * 32 + mi * 16 + j + 8 * lh;
                const int  c = wn * 64 + ni * 16 + lr;
                f(r, c, acc[mi][ni][j]);
            }
}

// ---------------------------------------------------------------------------
// Kernel 0: transpose + bf16-convert the weights.
//   WqkvT[e][d] = Wqkv[d][e]   (1536 x 512)
//   WoutT[e][d] = Wout[d][e]   (512 x 512)
// ---------------------------------------------------------------------------
__global__ __launch_bounds__(256) void k_prep(
    const float* __restrict__ Wqkv, const float* __restrict__ Wout,
    bf16_t* __restrict__ WqkvT, bf16_t* __restrict__ WoutT)
{
    int idx = blockIdx.x * 256 + threadIdx.x;
    if (idx < QKV_N * DIM) {
        int e = idx >> 9, d = idx & 511;
        WqkvT[idx] = (bf16_t)Wqkv[(long)d * QKV_N + e];
    }
    if (idx < DIM * DIM) {
        int e = idx >> 9, d = idx & 511;
        WoutT[idx] = (bf16_t)Wout[(long)d * DIM + e];
    }
}

// ---------------------------------------------------------------------------
// Kernel 1: QKV projection. X[16384x512] f32  @ WqkvT -> Q (x 1/64), Krows, Vt
//   Q, Krows : [B][N][512] bf16 row-major; Vt : [B][512][N] bf16 (transposed)
// ---------------------------------------------------------------------------
__global__ __launch_bounds__(256) void k_gemm_qkv(
    const float* __restrict__ X, const bf16_t* __restrict__ WqkvT,
    bf16_t* __restrict__ Q, bf16_t* __restrict__ Kr, bf16_t* __restrict__ Vt)
{
    __shared__ bf16_t As[128 * 32];
    __shared__ bf16_t Bs[128 * 32];
    const long row0 = (long)blockIdx.y * 128;
    const int  col0 = blockIdx.x * 128;
    v8f acc[2][4];
    gemm_block<true>(X + row0 * DIM, DIM, WqkvT + (long)col0 * DIM, DIM,
                     DIM, As, Bs, acc);
    epilogue_for_each(acc, [&](long r, int c, float val) {
        const long row = row0 + r;
        const int  e   = col0 + c;
        if (e < 512) {
            Q[row * 512 + e] = (bf16_t)(val * (1.0f / 64.0f)); // folded q*k scale
        } else if (e < 1024) {
            Kr[row * 512 + (e - 512)] = (bf16_t)val;
        } else {
            const long b = row >> 11;
            const long m = row & 2047;
            Vt[(b * 512 + (e - 1024)) * SEQ + m] = (bf16_t)val;
        }
    });
}

// ---------------------------------------------------------------------------
// Kernel 2: scores. Per batch: P[n][m] = Q[n][:] . K[m][:]   (bf16 out)
// ---------------------------------------------------------------------------
__global__ __launch_bounds__(256) void k_gemm_scores(
    const bf16_t* __restrict__ Q, const bf16_t* __restrict__ Kr,
    bf16_t* __restrict__ P)
{
    __shared__ bf16_t As[128 * 32];
    __shared__ bf16_t Bs[128 * 32];
    const int  b    = blockIdx.z;
    const long row0 = (long)blockIdx.y * 128;
    const int  col0 = blockIdx.x * 128;
    const bf16_t* Qb = Q  + (long)b * SEQ * DIM;
    const bf16_t* Kb = Kr + (long)b * SEQ * DIM;
    bf16_t*       Pb = P  + (long)b * SEQ * SEQ;
    v8f acc[2][4];
    gemm_block<false>(Qb + row0 * DIM, DIM, Kb + (long)col0 * DIM, DIM,
                      DIM, As, Bs, acc);
    epilogue_for_each(acc, [&](long r, int c, float val) {
        Pb[(row0 + r) * SEQ + (col0 + c)] = (bf16_t)val;
    });
}

// ---------------------------------------------------------------------------
// Kernel 3: row softmax over 2048 columns, in place on bf16, f32 statistics.
// One 256-thread block per row; 8 elements per thread.
// ---------------------------------------------------------------------------
__global__ __launch_bounds__(256) void k_softmax(bf16_t* __restrict__ P)
{
    const long row = blockIdx.x;
    bf16_t* p = P + row * SEQ;
    const int t = threadIdx.x, lane = t & 31, wid = t >> 5;

    float v[8];
    float mx = -3.0e38f;
#pragma unroll
    for (int i = 0; i < 8; ++i) {
        v[i] = (float)p[t + i * 256];
        mx = fmaxf(mx, v[i]);
    }
#pragma unroll
    for (int off = 16; off >= 1; off >>= 1)
        mx = fmaxf(mx, __shfl_xor(mx, off, 32));

    __shared__ float red[8];
    if (lane == 0) red[wid] = mx;
    __syncthreads();
#pragma unroll
    for (int w = 0; w < 8; ++w) mx = fmaxf(mx, red[w]);
    __syncthreads();

    float s = 0.0f;
#pragma unroll
    for (int i = 0; i < 8; ++i) { v[i] = __expf(v[i] - mx); s += v[i]; }
#pragma unroll
    for (int off = 16; off >= 1; off >>= 1)
        s += __shfl_xor(s, off, 32);
    if (lane == 0) red[wid] = s;
    __syncthreads();
    float tot = 0.0f;
#pragma unroll
    for (int w = 0; w < 8; ++w) tot += red[w];
    const float inv = 1.0f / tot;
#pragma unroll
    for (int i = 0; i < 8; ++i) p[t + i * 256] = (bf16_t)(v[i] * inv);
}

// ---------------------------------------------------------------------------
// Kernel 4: O = P @ V (contraction over m=2048), fused residual: Ox = O + x
// B operand = Vt[b] which is [d][m] row-major -> contraction contiguous.
// ---------------------------------------------------------------------------
__global__ __launch_bounds__(256) void k_gemm_pv(
    const bf16_t* __restrict__ P, const bf16_t* __restrict__ Vt,
    const float* __restrict__ X, bf16_t* __restrict__ Ox)
{
    __shared__ bf16_t As[128 * 32];
    __shared__ bf16_t Bs[128 * 32];
    const int  b    = blockIdx.z;
    const long row0 = (long)blockIdx.y * 128;   // query row within batch
    const int  col0 = blockIdx.x * 128;         // d column
    const bf16_t* Pb = P  + (long)b * SEQ * SEQ;
    const bf16_t* Vb = Vt + (long)b * DIM * SEQ;
    v8f acc[2][4];
    gemm_block<false>(Pb + row0 * SEQ, SEQ, Vb + (long)col0 * SEQ, SEQ,
                      SEQ, As, Bs, acc);
    epilogue_for_each(acc, [&](long r, int c, float val) {
        const long g = (long)b * SEQ + row0 + r;   // global row
        const int  d = col0 + c;
        Ox[g * DIM + d] = (bf16_t)(val + X[g * DIM + d]);
    });
}

// ---------------------------------------------------------------------------
// Kernel 5: output projection. Y = Ox @ WoutT + bout  (f32 out)
// ---------------------------------------------------------------------------
__global__ __launch_bounds__(256) void k_gemm_out(
    const bf16_t* __restrict__ Ox, const bf16_t* __restrict__ WoutT,
    const float* __restrict__ bout, float* __restrict__ Y)
{
    __shared__ bf16_t As[128 * 32];
    __shared__ bf16_t Bs[128 * 32];
    const long row0 = (long)blockIdx.y * 128;
    const int  col0 = blockIdx.x * 128;
    v8f acc[2][4];
    gemm_block<false>(Ox + row0 * DIM, DIM, WoutT + (long)col0 * DIM, DIM,
                      DIM, As, Bs, acc);
    epilogue_for_each(acc, [&](long r, int c, float val) {
        Y[(row0 + r) * DIM + (col0 + c)] = val + bout[col0 + c];
    });
}

// ---------------------------------------------------------------------------
// Host-side orchestration
// ---------------------------------------------------------------------------
extern "C" void kernel_launch(void* const* d_in, const int* in_sizes, int n_in,
                              void* d_out, int out_size, void* d_ws, size_t ws_size,
                              hipStream_t stream)
{
    (void)in_sizes; (void)n_in; (void)out_size; (void)ws_size;
    const float* x    = (const float*)d_in[0];   // [8,2048,512]
    const float* Wqkv = (const float*)d_in[1];   // [512,1536]
    const float* Wout = (const float*)d_in[2];   // [512,512]
    const float* bout = (const float*)d_in[3];   // [512]
    float*       out  = (float*)d_out;           // [8,2048,512]

    char* ws = (char*)d_ws;
    size_t off = 0;
    bf16_t* WqkvT = (bf16_t*)(ws + off); off += (size_t)QKV_N * DIM * 2;
    bf16_t* WoutT = (bf16_t*)(ws + off); off += (size_t)DIM * DIM * 2;
    bf16_t* Q     = (bf16_t*)(ws + off); off += (size_t)ROWS * DIM * 2;
    bf16_t* Kr    = (bf16_t*)(ws + off); off += (size_t)ROWS * DIM * 2;
    bf16_t* Vt    = (bf16_t*)(ws + off); off += (size_t)BATCH * DIM * SEQ * 2;
    bf16_t* P     = (bf16_t*)(ws + off); off += (size_t)BATCH * SEQ * SEQ * 2;
    bf16_t* Ox    = (bf16_t*)(ws + off); off += (size_t)ROWS * DIM * 2;

    // 0) weight transpose + bf16 convert
    k_prep<<<dim3((QKV_N * DIM + 255) / 256), dim3(256), 0, stream>>>(
        Wqkv, Wout, WqkvT, WoutT);

    // 1) QKV projection (x is f32, converted in-kernel)
    k_gemm_qkv<<<dim3(QKV_N / 128, ROWS / 128), dim3(256), 0, stream>>>(
        x, WqkvT, Q, Kr, Vt);

    // 2) scores = Q . K^T  (intermediate lives in L2: 64 MB bf16 < 192 MB)
    k_gemm_scores<<<dim3(SEQ / 128, SEQ / 128, BATCH), dim3(256), 0, stream>>>(
        Q, Kr, P);

    // 3) row softmax
    k_softmax<<<dim3(BATCH * SEQ), dim3(256), 0, stream>>>(P);

    // 4) O = P.V, fused residual add with x
    k_gemm_pv<<<dim3(DIM / 128, SEQ / 128, BATCH), dim3(256), 0, stream>>>(
        P, Vt, x, Ox);

    // 5) Y = (O + x) @ Wout + bout
    k_gemm_out<<<dim3(DIM / 128, ROWS / 128), dim3(256), 0, stream>>>(
        Ox, WoutT, bout, out);
}
